// TripletGATMeta_1554778161593
// MI455X (gfx1250) — matrix operand order repaired and checked by reading, not compiled, not run on hardware
//
#include <hip/hip_runtime.h>

typedef __attribute__((ext_vector_type(16))) _Float16 v16h;
typedef __attribute__((ext_vector_type(8)))  _Float16 v8h;
typedef __attribute__((ext_vector_type(8)))  float    v8f;

#define EMB   128
#define HC    1024   // HEADS*HID
#define HIDC  256

// ---------------------------------------------------------------------------
// helpers
// ---------------------------------------------------------------------------
static __device__ __forceinline__ void atomicMaxF(float* a, float v) {
  // works when *a initialized to -inf: positives via signed-int max,
  // negatives via unsigned-int min (more negative float = larger uint)
  if (v >= 0.f) atomicMax((int*)a, __float_as_int(v));
  else          atomicMin((unsigned int*)a, __float_as_uint(v));
}

// A fragment (16x32 f16, M rows): lane l holds row m0+(l&15);
// elements 0..7 -> K = kb + (l>>4)*8 + 0..7 ; elements 8..15 -> +16
static __device__ __forceinline__ v16h load_afrag(const _Float16* __restrict__ A,
                                                  int M, int K, int m0, int kb, int lane) {
  int m = m0 + (lane & 15); if (m >= M) m = M - 1;
  int half = lane >> 4;
  const _Float16* p = A + (size_t)m * K + kb + half * 8;
  v8h lo = *(const v8h*)(p);
  v8h hi = *(const v8h*)(p + 16);
  v16h a;
#pragma unroll
  for (int i = 0; i < 8; ++i) { a[i] = lo[i]; a[8 + i] = hi[i]; }
  return a;
}

// B fragment (32x16 f16) from pre-transposed BT[N][K]: lane l holds col n0+(l&15);
// elements 0..7 -> K = kb + (l>>4)*8 + 0..7 ; elements 8..15 -> +16
static __device__ __forceinline__ v16h load_bfrag(const _Float16* __restrict__ BT,
                                                  int K, int n0, int kb, int lane) {
  int n = n0 + (lane & 15);
  int half = lane >> 4;
  const _Float16* p = BT + (size_t)n * K + kb + half * 8;
  v8h lo = *(const v8h*)(p);
  v8h hi = *(const v8h*)(p + 16);
  v16h b;
#pragma unroll
  for (int i = 0; i < 8; ++i) { b[i] = lo[i]; b[8 + i] = hi[i]; }
  return b;
}

#define WMMA_F16(a, b, c) \
  __builtin_amdgcn_wmma_f32_16x16x32_f16(false, (a), false, (b), (short)0, (c), false, false)

// ---------------------------------------------------------------------------
// WMMA GEMM: C[M,N](f16) = act(A[M,K](f16) @ BT[N,K]^T + bias(f32))
// one wave -> 32x32 output tile (2x2 WMMA blocking: 8 b128 loads -> 4 WMMAs
// per K-step); K multiple of 32, N multiple of 32
// ---------------------------------------------------------------------------
__global__ void wmma_gemm_kernel(const _Float16* __restrict__ A,
                                 const _Float16* __restrict__ BT,
                                 const float* __restrict__ bias,
                                 _Float16* __restrict__ C,
                                 int M, int N, int K, int relu) {
  int lane = threadIdx.x & 31;
  int wave = threadIdx.x >> 5;
  int unit = blockIdx.x * (blockDim.x >> 5) + wave;
  int nt2 = N >> 5;
  int mt2 = (M + 31) >> 5;
  if (unit >= mt2 * nt2) return;            // wave-uniform: EXEC stays all-1 for WMMA
  int mt = unit / nt2, nt = unit % nt2;
  int m0 = mt << 5, n0 = nt << 5;

  v8f c00 = {}; v8f c01 = {}; v8f c10 = {}; v8f c11 = {};
  for (int kb = 0; kb < K; kb += 32) {
    v16h a0 = load_afrag(A, M, K, m0,      kb, lane);
    v16h a1 = load_afrag(A, M, K, m0 + 16, kb, lane);
    v16h b0 = load_bfrag(BT, K, n0,      kb, lane);
    v16h b1 = load_bfrag(BT, K, n0 + 16, kb, lane);
    c00 = WMMA_F16(a0, b0, c00);
    c01 = WMMA_F16(a0, b1, c01);
    c10 = WMMA_F16(a1, b0, c10);
    c11 = WMMA_F16(a1, b1, c11);
  }
  int col = lane & 15, half = lane >> 4;
  float bb0 = bias ? bias[n0 + col]      : 0.f;
  float bb1 = bias ? bias[n0 + 16 + col] : 0.f;
#pragma unroll
  for (int i = 0; i < 8; ++i) {
    int mA = m0 + i + half * 8;
    int mB = mA + 16;
    if (mA < M) {
      float v0 = c00[i] + bb0, v1 = c01[i] + bb1;
      if (relu) { v0 = v0 > 0.f ? v0 : 0.f; v1 = v1 > 0.f ? v1 : 0.f; }
      C[(size_t)mA * N + n0 + col]      = (_Float16)v0;
      C[(size_t)mA * N + n0 + 16 + col] = (_Float16)v1;
    }
    if (mB < M) {
      float v0 = c10[i] + bb0, v1 = c11[i] + bb1;
      if (relu) { v0 = v0 > 0.f ? v0 : 0.f; v1 = v1 > 0.f ? v1 : 0.f; }
      C[(size_t)mB * N + n0 + col]      = (_Float16)v0;
      C[(size_t)mB * N + n0 + 16 + col] = (_Float16)v1;
    }
  }
}

// ---------------------------------------------------------------------------
// elementwise / conversion kernels
// ---------------------------------------------------------------------------
__global__ void cvt_f16_kernel(const float* __restrict__ x, _Float16* __restrict__ y, long n) {
  long i = (long)blockIdx.x * blockDim.x + threadIdx.x;
  if (i < n) y[i] = (_Float16)x[i];
}

// W[K,N] f32 -> WT[N,K] f16
__global__ void transpose_cvt_kernel(const float* __restrict__ W, _Float16* __restrict__ WT,
                                     int K, int N) {
  long i = (long)blockIdx.x * blockDim.x + threadIdx.x;
  if (i >= (long)K * N) return;
  int k = (int)(i / N), n = (int)(i % N);
  WT[(size_t)n * K + k] = (_Float16)W[i];
}

__global__ void fill_f32_kernel(float* __restrict__ p, float v, long n) {
  long i = (long)blockIdx.x * blockDim.x + threadIdx.x;
  if (i < n) p[i] = v;
}

__global__ void elu_cvt_kernel(const float* __restrict__ H, _Float16* __restrict__ H16, long n) {
  long i = (long)blockIdx.x * blockDim.x + threadIdx.x;
  if (i >= n) return;
  float v = H[i];
  v = v > 0.f ? v : (__expf(v) - 1.f);
  H16[i] = (_Float16)v;
}

__global__ void mul_f16_kernel(const _Float16* __restrict__ X, const _Float16* __restrict__ Y,
                               _Float16* __restrict__ Z, long n) {
  long i = (long)blockIdx.x * blockDim.x + threadIdx.x;
  if (i < n) Z[i] = (_Float16)((float)X[i] * (float)Y[i]);
}

// CAT[b] = [HA[ia[b]] (256) | HB[ib[b]] (256)]   (one wave per row, b128 copies)
__global__ void gather_concat_kernel(const _Float16* __restrict__ HA, const _Float16* __restrict__ HB,
                                     const int* __restrict__ ia, const int* __restrict__ ib,
                                     _Float16* __restrict__ CAT, int B) {
  int w = (blockIdx.x * blockDim.x + threadIdx.x) >> 5;
  int lane = threadIdx.x & 31;
  if (w >= B) return;
  const v8h* pa = (const v8h*)(HA + (size_t)ia[w] * HIDC);
  const v8h* pb = (const v8h*)(HB + (size_t)ib[w] * HIDC);
  v8h* pc = (v8h*)(CAT + (size_t)w * (2 * HIDC));
  pc[lane]      = pa[lane];
  pc[32 + lane] = pb[lane];
}

// out[b] = sum_c X[b,c]*w[c] + bias[0]   (wave per row)
__global__ void dot256_kernel(const _Float16* __restrict__ X, const float* __restrict__ w,
                              const float* __restrict__ bias, float* __restrict__ out, int B) {
  int wv = (blockIdx.x * blockDim.x + threadIdx.x) >> 5;
  int lane = threadIdx.x & 31;
  if (wv >= B) return;
  v8h x = ((const v8h*)(X + (size_t)wv * HIDC))[lane];
  float acc = 0.f;
#pragma unroll
  for (int i = 0; i < 8; ++i) acc += (float)x[i] * w[lane * 8 + i];
#pragma unroll
  for (int off = 16; off > 0; off >>= 1) acc += __shfl_xor(acc, off, 32);
  if (lane == 0) out[wv] = acc + bias[0];
}

// ---------------------------------------------------------------------------
// edge kernels (one wave per edge; lane covers 32 of 1024 channels -> one head)
// ---------------------------------------------------------------------------
__global__ void edge_score_kernel(const _Float16* __restrict__ XL, const _Float16* __restrict__ XR,
                                  const int* __restrict__ src, const int* __restrict__ dst,
                                  const float* __restrict__ att, float* __restrict__ score,
                                  float* __restrict__ smax, int E) {
  int e = (blockIdx.x * blockDim.x + threadIdx.x) >> 5;
  int lane = threadIdx.x & 31;
  if (e >= E) return;
  int s = src[e], d = dst[e];
  const _Float16* xl = XL + (size_t)s * HC + lane * 32;
  const _Float16* xr = XR + (size_t)d * HC + lane * 32;
  const float* at = att + lane * 32;
  float acc = 0.f;
#pragma unroll
  for (int i0 = 0; i0 < 32; i0 += 8) {
    v8h a = *(const v8h*)(xl + i0);
    v8h b = *(const v8h*)(xr + i0);
#pragma unroll
    for (int i = 0; i < 8; ++i) {
      float v = (float)a[i] + (float)b[i];
      v = v > 0.f ? v : 0.2f * v;               // leaky_relu(0.2)
      acc += v * at[i0 + i];
    }
  }
  acc += __shfl_xor(acc, 1, 32);
  acc += __shfl_xor(acc, 2, 32);
  acc += __shfl_xor(acc, 4, 32);
  if ((lane & 7) == 0) {
    int h = lane >> 3;
    score[(size_t)e * 4 + h] = acc;
    atomicMaxF(&smax[(size_t)d * 4 + h], acc);
  }
}

__global__ void edge_exp_kernel(const float* __restrict__ score, const float* __restrict__ smax,
                                const int* __restrict__ dst, float* __restrict__ ex,
                                float* __restrict__ denom, int E) {
  long t = (long)blockIdx.x * blockDim.x + threadIdx.x;
  if (t >= (long)E * 4) return;
  int e = (int)(t >> 2), h = (int)(t & 3);
  int d = dst[e];
  float v = __expf(score[t] - smax[(size_t)d * 4 + h]);
  ex[t] = v;
  atomicAdd(&denom[(size_t)d * 4 + h], v);
}

__global__ void edge_agg_kernel(const _Float16* __restrict__ XL,
                                const int* __restrict__ src, const int* __restrict__ dst,
                                const float* __restrict__ ex, const float* __restrict__ denom,
                                float* __restrict__ AG, int E) {
  int e = (blockIdx.x * blockDim.x + threadIdx.x) >> 5;
  int lane = threadIdx.x & 31;
  if (e >= E) return;
  int s = src[e], d = dst[e];
  int h = lane >> 3;
  float alpha = ex[(size_t)e * 4 + h] / (denom[(size_t)d * 4 + h] + 1e-16f);
  const _Float16* xl = XL + (size_t)s * HC + lane * 32;
  float* ag = AG + (size_t)d * HC + lane * 32;
#pragma unroll
  for (int i0 = 0; i0 < 32; i0 += 8) {
    v8h a = *(const v8h*)(xl + i0);
#pragma unroll
    for (int i = 0; i < 8; ++i) atomicAdd(&ag[i0 + i], alpha * (float)a[i]);
  }
}

// head-mean + bout, scaled accumulate into destination hidden state
__global__ void finalize_rel_kernel(const float* __restrict__ AG, const float* __restrict__ bout,
                                    float* __restrict__ Hacc, float scale, int Nd) {
  long t = (long)blockIdx.x * blockDim.x + threadIdx.x;
  if (t >= (long)Nd * HIDC) return;
  int n = (int)(t >> 8), c = (int)(t & 255);
  const float* ag = AG + (size_t)n * HC;
  float v = 0.25f * (ag[c] + ag[256 + c] + ag[512 + c] + ag[768 + c]) + bout[c];
  Hacc[t] += scale * v;
}

// ---------------------------------------------------------------------------
// host
// ---------------------------------------------------------------------------
static inline size_t amax(size_t a, size_t b) { return a > b ? a : b; }

extern "C" void kernel_launch(void* const* d_in, const int* in_sizes, int n_in,
                              void* d_out, int out_size, void* d_ws, size_t ws_size,
                              hipStream_t stream) {
  // ---- inputs ----
  const float* emb_pep = (const float*)d_in[0];
  const float* emb_mhc = (const float*)d_in[1];
  const float* emb_tcr = (const float*)d_in[2];
  const float* Wl   = (const float*)d_in[3];
  const float* bl   = (const float*)d_in[4];
  const float* Wr   = (const float*)d_in[5];
  const float* br   = (const float*)d_in[6];
  const float* att  = (const float*)d_in[7];
  const float* bout = (const float*)d_in[8];
  const float* fpm_W1 = (const float*)d_in[9];  const float* fpm_b1 = (const float*)d_in[10];
  const float* fpm_W2 = (const float*)d_in[11]; const float* fpm_b2 = (const float*)d_in[12];
  const float* fmt_W1 = (const float*)d_in[13]; const float* fmt_b1 = (const float*)d_in[14];
  const float* fmt_W2 = (const float*)d_in[15]; const float* fmt_b2 = (const float*)d_in[16];
  const float* fdmf_W1 = (const float*)d_in[17]; const float* fdmf_b1 = (const float*)d_in[18];
  const float* fdmf_W2 = (const float*)d_in[19]; const float* fdmf_b2 = (const float*)d_in[20];
  const float* wpm_W = (const float*)d_in[21];   const float* wpm_b = (const float*)d_in[22];
  const int* pep_idx = (const int*)d_in[33];
  const int* mhc_idx = (const int*)d_in[34];
  const int* tcr_idx = (const int*)d_in[35];

  const int n_pep = in_sizes[0] / EMB;
  const int n_mhc = in_sizes[1] / EMB;
  const int n_tcr = in_sizes[2] / EMB;
  const int B = in_sizes[33];
  int nmax = n_pep; if (n_mhc > nmax) nmax = n_mhc; if (n_tcr > nmax) nmax = n_tcr;
  int Emax = 0; for (int r = 0; r < 5; ++r) if (in_sizes[23 + 2 * r] > Emax) Emax = in_sizes[23 + 2 * r];

  // ---- workspace bump allocator ----
  char* w = (char*)d_ws; size_t off = 0;
  auto alloc = [&](size_t bytes) -> void* {
    void* p = w + off; off += (bytes + 255) & ~(size_t)255; return p;
  };
  _Float16* e16_pep = (_Float16*)alloc((size_t)n_pep * EMB * 2);
  _Float16* e16_mhc = (_Float16*)alloc((size_t)n_mhc * EMB * 2);
  _Float16* e16_tcr = (_Float16*)alloc((size_t)n_tcr * EMB * 2);
  _Float16* WlT = (_Float16*)alloc((size_t)5 * HC * EMB * 2);
  _Float16* WrT = (_Float16*)alloc((size_t)5 * HC * EMB * 2);
  _Float16* fpmW1T = (_Float16*)alloc((size_t)HIDC * 512 * 2);
  _Float16* fpmW2T = (_Float16*)alloc((size_t)HIDC * HIDC * 2);
  _Float16* fmtW1T = (_Float16*)alloc((size_t)HIDC * 512 * 2);
  _Float16* fmtW2T = (_Float16*)alloc((size_t)HIDC * HIDC * 2);
  _Float16* fdmfW1T = (_Float16*)alloc((size_t)HIDC * HIDC * 2);
  float* score = (float*)alloc((size_t)Emax * 4 * 4);
  float* ex    = (float*)alloc((size_t)Emax * 4 * 4);
  float* smax  = (float*)alloc((size_t)nmax * 4 * 4);
  float* denom = (float*)alloc((size_t)nmax * 4 * 4);
  float* Hpep = (float*)alloc((size_t)n_pep * HIDC * 4);
  float* Hmhc = (float*)alloc((size_t)n_mhc * HIDC * 4);
  float* Htcr = (float*)alloc((size_t)n_tcr * HIDC * 4);
  _Float16* H16pep = (_Float16*)alloc((size_t)n_pep * HIDC * 2);
  _Float16* H16mhc = (_Float16*)alloc((size_t)n_mhc * HIDC * 2);
  _Float16* H16tcr = (_Float16*)alloc((size_t)n_tcr * HIDC * 2);
  // big aliased regions: phase A (XL/XR/AG) vs phase C (MLP buffers)
  size_t szProj = (size_t)nmax * HC * 2;        // f16 [nmax,1024]
  size_t szVec  = (size_t)B * HIDC * 2;         // f16 [B,256]
  size_t szR1 = amax(szProj, 2 * szVec);
  size_t szR3 = amax((size_t)nmax * HC * 4, amax((size_t)B * 512 * 2, szVec));
  char* R1 = (char*)alloc(szR1);
  char* R2 = (char*)alloc(szR1);
  char* R3 = (char*)alloc(szR3);
  _Float16* XL = (_Float16*)R1;
  _Float16* XR = (_Float16*)R2;
  float*    AG = (float*)R3;
  _Float16* HIDbuf = (_Float16*)R1;
  _Float16* VPM = (_Float16*)(R1 + szVec);
  _Float16* VMT = (_Float16*)R2;
  _Float16* Zb  = (_Float16*)(R2 + szVec);
  _Float16* CAT = (_Float16*)R3;
  _Float16* HID2 = (_Float16*)R3;
  (void)ws_size; (void)n_in; (void)out_size;

  const int TB = 256;
  auto blks = [&](long n) { return (int)((n + TB - 1) / TB); };

  // ---- phase 0: conversions ----
  cvt_f16_kernel<<<blks((long)n_pep * EMB), TB, 0, stream>>>(emb_pep, e16_pep, (long)n_pep * EMB);
  cvt_f16_kernel<<<blks((long)n_mhc * EMB), TB, 0, stream>>>(emb_mhc, e16_mhc, (long)n_mhc * EMB);
  cvt_f16_kernel<<<blks((long)n_tcr * EMB), TB, 0, stream>>>(emb_tcr, e16_tcr, (long)n_tcr * EMB);
  for (int r = 0; r < 5; ++r) {
    transpose_cvt_kernel<<<blks((long)EMB * HC), TB, 0, stream>>>(Wl + (size_t)r * EMB * HC, WlT + (size_t)r * HC * EMB, EMB, HC);
    transpose_cvt_kernel<<<blks((long)EMB * HC), TB, 0, stream>>>(Wr + (size_t)r * EMB * HC, WrT + (size_t)r * HC * EMB, EMB, HC);
  }
  transpose_cvt_kernel<<<blks((long)512 * HIDC), TB, 0, stream>>>(fpm_W1, fpmW1T, 512, HIDC);
  transpose_cvt_kernel<<<blks((long)HIDC * HIDC), TB, 0, stream>>>(fpm_W2, fpmW2T, HIDC, HIDC);
  transpose_cvt_kernel<<<blks((long)512 * HIDC), TB, 0, stream>>>(fmt_W1, fmtW1T, 512, HIDC);
  transpose_cvt_kernel<<<blks((long)HIDC * HIDC), TB, 0, stream>>>(fmt_W2, fmtW2T, HIDC, HIDC);
  transpose_cvt_kernel<<<blks((long)HIDC * HIDC), TB, 0, stream>>>(fdmf_W1, fdmfW1T, HIDC, HIDC);

  hipMemsetAsync(Hpep, 0, (size_t)n_pep * HIDC * 4, stream);
  hipMemsetAsync(Hmhc, 0, (size_t)n_mhc * HIDC * 4, stream);
  hipMemsetAsync(Htcr, 0, (size_t)n_tcr * HIDC * 4, stream);

  auto gemm = [&](const _Float16* A, const _Float16* BT, const float* bias,
                  _Float16* C, int M, int N, int K, int relu) {
    long units = (long)((M + 31) / 32) * (N / 32);
    wmma_gemm_kernel<<<(int)((units + 7) / 8), 256, 0, stream>>>(A, BT, bias, C, M, N, K, relu);
  };

  // ---- phase A: 5 GATv2 relations ----
  const _Float16* srcEmb[5] = {e16_pep, e16_mhc, e16_pep, e16_mhc, e16_tcr};
  const int srcN[5]         = {n_pep,   n_mhc,   n_pep,   n_mhc,   n_tcr};
  const _Float16* dstEmb[5] = {e16_mhc, e16_tcr, e16_tcr, e16_pep, e16_pep};
  const int dstN[5]         = {n_mhc,   n_tcr,   n_tcr,   n_pep,   n_pep};
  float* Hdst[5]            = {Hmhc,    Htcr,    Htcr,    Hpep,    Hpep};
  const float scale[5]      = {1.0f,    0.5f,    0.5f,    0.5f,    0.5f};

  for (int r = 0; r < 5; ++r) {
    const int* src = (const int*)d_in[23 + 2 * r];
    const int* dst = (const int*)d_in[24 + 2 * r];
    const int E = in_sizes[23 + 2 * r];
    const int Ns = srcN[r], Nd = dstN[r];

    gemm(srcEmb[r], WlT + (size_t)r * HC * EMB, bl + (size_t)r * HC, XL, Ns, HC, EMB, 0);
    gemm(dstEmb[r], WrT + (size_t)r * HC * EMB, br + (size_t)r * HC, XR, Nd, HC, EMB, 0);

    fill_f32_kernel<<<blks((long)Nd * 4), TB, 0, stream>>>(smax, -__builtin_huge_valf(), (long)Nd * 4);
    hipMemsetAsync(denom, 0, (size_t)Nd * 4 * 4, stream);
    hipMemsetAsync(AG, 0, (size_t)Nd * HC * 4, stream);

    int eblk = (E + (TB / 32) - 1) / (TB / 32);
    edge_score_kernel<<<eblk, TB, 0, stream>>>(XL, XR, src, dst, att + (size_t)r * HC, score, smax, E);
    edge_exp_kernel<<<blks((long)E * 4), TB, 0, stream>>>(score, smax, dst, ex, denom, E);
    edge_agg_kernel<<<eblk, TB, 0, stream>>>(XL, src, dst, ex, denom, AG, E);
    finalize_rel_kernel<<<blks((long)Nd * HIDC), TB, 0, stream>>>(AG, bout + (size_t)r * HIDC, Hdst[r], scale[r], Nd);
  }

  // ---- phase B: ELU + f16 ----
  elu_cvt_kernel<<<blks((long)n_pep * HIDC), TB, 0, stream>>>(Hpep, H16pep, (long)n_pep * HIDC);
  elu_cvt_kernel<<<blks((long)n_mhc * HIDC), TB, 0, stream>>>(Hmhc, H16mhc, (long)n_mhc * HIDC);
  elu_cvt_kernel<<<blks((long)n_tcr * HIDC), TB, 0, stream>>>(Htcr, H16tcr, (long)n_tcr * HIDC);

  // ---- phase C: MLPs ----
  float* pm_logit = (float*)d_out;
  float* pmt_logit = (float*)d_out + B;
  int gblk = (B + (TB / 32) - 1) / (TB / 32);

  // v_pm = mlp2([hp,hm])
  gather_concat_kernel<<<gblk, TB, 0, stream>>>(H16pep, H16mhc, pep_idx, mhc_idx, CAT, B);
  gemm(CAT, fpmW1T, fpm_b1, HIDbuf, B, HIDC, 512, 1);
  gemm(HIDbuf, fpmW2T, fpm_b2, VPM, B, HIDC, HIDC, 0);
  dot256_kernel<<<gblk, TB, 0, stream>>>(VPM, wpm_W, wpm_b, pm_logit, B);

  // v_mt = mlp2([hm,ht])
  gather_concat_kernel<<<gblk, TB, 0, stream>>>(H16mhc, H16tcr, mhc_idx, tcr_idx, CAT, B);
  gemm(CAT, fmtW1T, fmt_b1, HIDbuf, B, HIDC, 512, 1);
  gemm(HIDbuf, fmtW2T, fmt_b2, VMT, B, HIDC, HIDC, 0);

  // pmt = mlp2(v_pm * v_mt)
  mul_f16_kernel<<<blks((long)B * HIDC), TB, 0, stream>>>(VPM, VMT, Zb, (long)B * HIDC);
  gemm(Zb, fdmfW1T, fdmf_b1, HID2, B, HIDC, HIDC, 1);
  dot256_kernel<<<gblk, TB, 0, stream>>>(HID2, fdmf_W2, fdmf_b2, pmt_logit, B);
}
